// UpsamplingBaselineNetwork_15564961481346
// MI455X (gfx1250) — matrix operand order repaired and checked by reading, not compiled, not run on hardware
//
#include <hip/hip_runtime.h>

// ---------------------------------------------------------------------------
// CDNA5 (gfx1250) SR network: all convs are implicit GEMM on
// V_WMMA_F32_16X16X32_F16 (f16 in, fp32 acc). wave32, 256-thread blocks.
// B staging: async global->LDS b128 from pre-converted padded f16 weights.
// A staging: register-pipelined (next K-slab fetched during WMMA), packed
// b128 LDS stores. Epilogues emit row-contiguous b128 stores / vector RMW.
// ---------------------------------------------------------------------------

typedef __attribute__((ext_vector_type(16))) _Float16 v16h;
typedef __attribute__((ext_vector_type(8)))  _Float16 v8h;
typedef __attribute__((ext_vector_type(8)))  float    v8f;
typedef __attribute__((ext_vector_type(4)))  float    v4f;
typedef __attribute__((ext_vector_type(4)))  int      v4i;

#if defined(__AMDGCN__) && __has_builtin(__builtin_amdgcn_global_load_async_to_lds_b128)
#define USE_ASYNC_LDS 1
typedef const __attribute__((address_space(1))) void* as1cp;
typedef __attribute__((address_space(3))) void*       as3p;
typedef __attribute__((address_space(1))) v4i*        g_v4i_p;
typedef __attribute__((address_space(3))) v4i*        l_v4i_p;
#if __has_builtin(__builtin_amdgcn_s_wait_asynccnt)
#define WAIT_ASYNC() __builtin_amdgcn_s_wait_asynccnt(0)
#else
#define WAIT_ASYNC() asm volatile("s_wait_asynccnt 0x0" ::: "memory")
#endif
#else
#define USE_ASYNC_LDS 0
#define WAIT_ASYNC()
#endif

constexpr int MODE_STORE      = 0;  // out = acc + bias
constexpr int MODE_RELU_STORE = 1;  // out = relu(acc + bias)
constexpr int MODE_ADD        = 2;  // out += acc + bias (Tout == float)
constexpr int MODE_PS2_RELU   = 3;  // pixel_shuffle(2) store of relu(acc+bias)

constexpr int LSTR = 40;  // LDS row stride in halfs: 80B, 16B-aligned, conflict-free

// Tin/Tout: float or _Float16. wgt16: padded f16 blob, row stride Kpad halfs.
template <typename Tin, typename Tout, int MT, int NT, int WM, int NSUB,
          int MODE, bool RELU_IN>
__global__ __launch_bounds__(256) void conv3x3_wmma(
    const Tin* __restrict__ in, const _Float16* __restrict__ wgt16, int Kpad,
    const float* __restrict__ bias, Tout* __restrict__ out,
    int B, int Cin, int H, int W, int Cout)
{
  __shared__ __align__(16) _Float16 As[MT * LSTR];
  __shared__ __align__(16) _Float16 Bs[NT * LSTR];

  constexpr int NEL = (MT * 32) / 256;   // contiguous k-halfs per thread per slab
  constexpr int LOG2MT = (MT == 64) ? 6 : 7;
  constexpr int BCH = NT * 4;            // 16-byte B chunks per slab

  const int K    = Cin * 9;
  const int HW   = H * W;
  const int tid  = threadIdx.x;
  const int lane = tid & 31;
  const int wave = tid >> 5;
  const int wm   = (wave % WM) << 4;
  const int wn   = (wave / WM) * (NSUB << 4);
  const int hi   = (lane >> 4) & 1;
  const int lr   = lane & 15;

  const int mBase = blockIdx.x * MT;     // flat (b*HW + pixel) base
  const int nBase = blockIdx.y * NT;     // output-channel base

  // ---- per-thread A-staging decode, hoisted out of the K loop ----
  const int mA   = tid & (MT - 1);
  const int kkB  = (tid >> LOG2MT) * NEL;   // contiguous chunk base
  const int pixA = mBase + mA;
  const int bA   = pixA / HW;
  const int pA   = pixA - bA * HW;
  const int yA   = pA / W;
  const int xA   = pA - yA * W;
  const int planeA = bA * Cin;

  v8f acc[NSUB];
#pragma unroll
  for (int s = 0; s < NSUB; ++s)
#pragma unroll
    for (int i = 0; i < 8; ++i) acc[s][i] = 0.0f;

  const int numK = Kpad >> 5;
  float ra[NEL];

  // fetch one K-slab of A into registers (clamped address + select, no branches)
  auto fetchA = [&](int k0) {
#pragma unroll
    for (int i = 0; i < NEL; ++i) {
      const int kg = k0 + kkB + i;
      const unsigned c = (unsigned)kg / 9u;
      const unsigned t = (unsigned)kg - 9u * c;
      const int r  = (int)(t / 3u) - 1;
      const int sx = (int)(t - 3u * (t / 3u)) - 1;
      const int y  = yA + r;
      const int x  = xA + sx;
      const bool ok = (kg < K) & (y >= 0) & (y < H) & (x >= 0) & (x < W);
      const int addr = ok ? (((planeA + (int)c) * H + y) * W + x) : 0;
      float v = (float)in[addr];
      v = ok ? v : 0.0f;
      if (RELU_IN) v = fmaxf(v, 0.0f);
      ra[i] = v;
    }
  };

  fetchA(0);

  for (int ks = 0; ks < numK; ++ks) {
    const int k0 = ks << 5;

    // ---- B slab: async 16B-chunk copy from padded f16 weight blob ----
    if (BCH == 256 || tid < BCH) {
      const int n  = tid >> 2;
      const int cw = tid & 3;
      const _Float16* gsrc = wgt16 + (size_t)(nBase + n) * Kpad + k0 + cw * 8;
      _Float16* ldst = Bs + n * LSTR + cw * 8;
#if USE_ASYNC_LDS
      __builtin_amdgcn_global_load_async_to_lds_b128(
          (g_v4i_p)(as1cp)(const void*)gsrc, (l_v4i_p)(as3p)(void*)ldst, 0, 0);
#else
      *(v8h*)ldst = *(const v8h*)gsrc;
#endif
    }

    // ---- A slab: registers -> LDS, packed b128 stores ----
#pragma unroll
    for (int ch = 0; ch < NEL / 8; ++ch) {
      v8h pk;
#pragma unroll
      for (int i = 0; i < 8; ++i) pk[i] = (_Float16)ra[ch * 8 + i];
      *(v8h*)(As + mA * LSTR + kkB + ch * 8) = pk;
    }

    WAIT_ASYNC();
    __syncthreads();

    // issue next slab's global loads early (overlap with WMMA)
    if (ks + 1 < numK) fetchA((ks + 1) << 5);

    // ---- fragments (ISA 7.12.2 layouts) + WMMA ----
    const _Float16* arow = As + (wm + lr) * LSTR;
    const v8h a0 = *(const v8h*)(arow + (hi << 3));
    const v8h a1 = *(const v8h*)(arow + 16 + (hi << 3));
    v16h a;
#pragma unroll
    for (int i = 0; i < 8; ++i) { a[i] = a0[i]; a[8 + i] = a1[i]; }

#pragma unroll
    for (int s = 0; s < NSUB; ++s) {
      const _Float16* brow = Bs + (wn + (s << 4) + lr) * LSTR;
      const v8h b0 = *(const v8h*)(brow + (hi << 3));
      const v8h b1 = *(const v8h*)(brow + 16 + (hi << 3));
      v16h b;
#pragma unroll
      for (int i = 0; i < 8; ++i) { b[i] = b0[i]; b[8 + i] = b1[i]; }
      acc[s] = __builtin_amdgcn_wmma_f32_16x16x32_f16(
          false, a, false, b, (short)0, acc[s], false, false);
    }
    __syncthreads();
  }

  // ---- epilogue ----
  // C layout: VGPR vv -> M = wm + hi*8 + vv, N = lr. For a fixed lane the 8
  // accumulator elements are 8 consecutive pixels in one row -> vector stores.
  const int p0 = mBase + wm + (hi << 3);
  const int b  = p0 / HW;
  const int pp = p0 - b * HW;
  const int y  = pp / W;
  const int x  = pp - y * W;

#pragma unroll
  for (int s = 0; s < NSUB; ++s) {
    const int ng = nBase + wn + (s << 4) + lr;
    if (ng >= Cout) continue;
    const float bv = bias[ng];
    float vals[8];
#pragma unroll
    for (int vv = 0; vv < 8; ++vv) {
      float v = acc[s][vv] + bv;
      if constexpr (MODE == MODE_RELU_STORE || MODE == MODE_PS2_RELU)
        v = fmaxf(v, 0.0f);
      vals[vv] = v;
    }

    if constexpr (MODE == MODE_PS2_RELU) {
      const int co = ng >> 2, ry = (ng >> 1) & 1, rx = ng & 1;
      const int Cp = Cout >> 2;
      const int rowbase = ((b * Cp + co) * (2 * H) + (2 * y + ry)) * (2 * W) + rx;
#pragma unroll
      for (int vv = 0; vv < 8; ++vv)
        out[rowbase + 2 * (x + vv)] = (Tout)vals[vv];
    } else {
      const int idx0 = ((b * Cout + ng) * H + y) * W + x;
      if constexpr (MODE == MODE_ADD) {
        // Tout == float for all ADD uses: vector read-modify-write
        v4f* po = (v4f*)((float*)out + idx0);
        v4f lo = po[0], hv = po[1];
#pragma unroll
        for (int i = 0; i < 4; ++i) { lo[i] += vals[i]; hv[i] += vals[4 + i]; }
        po[0] = lo; po[1] = hv;
      } else if constexpr (sizeof(Tout) == 2) {
        v8h pk;
#pragma unroll
        for (int i = 0; i < 8; ++i) pk[i] = (_Float16)vals[i];
        *(v8h*)((_Float16*)out + idx0) = pk;
      } else {
        v4f lo, hv;
#pragma unroll
        for (int i = 0; i < 4; ++i) { lo[i] = vals[i]; hv[i] = vals[4 + i]; }
        v4f* po = (v4f*)((float*)out + idx0);
        po[0] = lo; po[1] = hv;
      }
    }
  }
}

// ---------------------------------------------------------------------------
// Weight pre-convert: fp32 [Cout,K] -> f16 [Npad,Kpad], zero-padded tails.
// ---------------------------------------------------------------------------
__global__ __launch_bounds__(256) void cvt_weights(
    const float* __restrict__ w, _Float16* __restrict__ o,
    int Cout, int K, int Kpad, int total)
{
  const int i = blockIdx.x * 256 + threadIdx.x;
  if (i >= total) return;
  const int n = i / Kpad;
  const int k = i - n * Kpad;
  const float v = (n < Cout && k < K) ? w[n * K + k] : 0.0f;
  o[i] = (_Float16)v;
}

// ---------------------------------------------------------------------------
// Dynamic local filtering upsampler (scale 4, 5x5), fused normalization:
//   out[c][s] = sum_t p[c][t]*k[t][s] - (ksum[s]-1)*(1/25)*psum[c]
// kern is the f16 up_kernel tensor [4,400,128,128].
// ---------------------------------------------------------------------------
__global__ __launch_bounds__(256) void local_conv_us_kernel(
    const float* __restrict__ x, const _Float16* __restrict__ kern,
    float* __restrict__ out)
{
  const int t = blockIdx.x * blockDim.x + threadIdx.x;
  if (t >= 4 * 128 * 128) return;
  const int HW = 128 * 128;
  const int b = t >> 14;
  const int p = t & (HW - 1);
  const int h = p >> 7;
  const int w = p & 127;

  float acc[3][16];
  float psum[3] = {0.f, 0.f, 0.f};
  float ksum[16];
#pragma unroll
  for (int c = 0; c < 3; ++c)
#pragma unroll
    for (int s = 0; s < 16; ++s) acc[c][s] = 0.f;
#pragma unroll
  for (int s = 0; s < 16; ++s) ksum[s] = 0.f;

  const _Float16* kb = kern + (size_t)b * 400 * HW + p;
  for (int tap = 0; tap < 25; ++tap) {
    const int r  = tap / 5 - 2;
    const int sx = tap - (tap / 5) * 5 - 2;
    const int iy = h + r, ix = w + sx;
    const bool ok = (iy >= 0) & (iy < 128) & (ix >= 0) & (ix < 128);
    float pv[3];
#pragma unroll
    for (int c = 0; c < 3; ++c) {
      const int a = ok ? (((b * 3 + c) << 14) + (iy << 7) + ix) : 0;
      const float v = x[a];
      pv[c] = ok ? v : 0.f;
    }
    float kv[16];
#pragma unroll
    for (int s = 0; s < 16; ++s) kv[s] = (float)kb[(size_t)(tap * 16 + s) * HW];
#pragma unroll
    for (int s = 0; s < 16; ++s) ksum[s] += kv[s];
#pragma unroll
    for (int c = 0; c < 3; ++c) {
      psum[c] += pv[c];
#pragma unroll
      for (int s = 0; s < 16; ++s) acc[c][s] = fmaf(pv[c], kv[s], acc[c][s]);
    }
  }
  // 4 consecutive x-outputs per (c, py) -> one v4f store each
#pragma unroll
  for (int c = 0; c < 3; ++c)
#pragma unroll
    for (int py = 0; py < 4; ++py) {
      v4f o;
#pragma unroll
      for (int px = 0; px < 4; ++px) {
        const int s = py * 4 + px;
        o[px] = acc[c][s] - (ksum[s] - 1.0f) * 0.04f * psum[c];
      }
      *(v4f*)(out + ((b * 3 + c) * 512 + (4 * h + py)) * 512 + 4 * w) = o;
    }
}

// ---------------------------------------------------------------------------
extern "C" void kernel_launch(void* const* d_in, const int* in_sizes, int n_in,
                              void* d_out, int out_size, void* d_ws, size_t ws_size,
                              hipStream_t stream)
{
  const float* x    = (const float*)d_in[0];
  const float* w_in = (const float*)d_in[1];
  const float* b_in = (const float*)d_in[2];
  const float* rw1  = (const float*)d_in[3];
  const float* rb1  = (const float*)d_in[4];
  const float* rw2  = (const float*)d_in[5];
  const float* rb2  = (const float*)d_in[6];
  const float* r_w1 = (const float*)d_in[7];
  const float* r_b1 = (const float*)d_in[8];
  const float* r_w2 = (const float*)d_in[9];
  const float* r_b2 = (const float*)d_in[10];
  const float* r_w3 = (const float*)d_in[11];
  const float* r_b3 = (const float*)d_in[12];
  const float* k_w1 = (const float*)d_in[13];
  const float* k_b1 = (const float*)d_in[14];
  const float* k_w2 = (const float*)d_in[15];
  const float* k_b2 = (const float*)d_in[16];
  float* out = (float*)d_out;

  // Workspace layout (bytes):
  char* ws = (char*)d_ws;
  float*    F    = (float*)(ws);                     // [4,64,128,128] f32   16 MiB
  _Float16* T0   = (_Float16*)(ws + 16777216ull);    // [4,64,128,128] f16    8 MiB
  _Float16* KACT = (_Float16*)(ws + 25165824ull);    // [4,128,128,128] f16 / r1 PS out [4,32,256,256] f16  16 MiB
  _Float16* UPK  = (_Float16*)(ws + 41943040ull);    // [4,400,128,128] f16 / r2 PS out [4,16,512,512] f16  50 MiB
  _Float16* WG   = (_Float16*)(ws + 94371840ull);    // padded f16 weights, ~3 MiB

  // f16 weight blobs (offsets in halfs)
  _Float16* w_in16 = WG;            // [64 ,  32]
  _Float16* rw1_16 = WG + 2048;     // [768, 576] (12 blocks x 64 rows)
  _Float16* rw2_16 = WG + 444416;   // [768, 576]
  _Float16* r1_16  = WG + 886784;   // [128, 576]
  _Float16* r2_16  = WG + 960512;   // [64 , 288]
  _Float16* r3_16  = WG + 978944;   // [16 , 160]
  _Float16* k1_16  = WG + 981504;   // [128, 576]
  _Float16* k2_16  = WG + 1055232;  // [448,1152]

  const dim3 blk(256);
  auto cvtw = [&](const float* w, _Float16* o, int Cout, int K, int Kpad, int Npad) {
    const int total = Npad * Kpad;
    cvt_weights<<<dim3((total + 255) / 256), blk, 0, stream>>>(w, o, Cout, K, Kpad, total);
  };
  cvtw(w_in, w_in16, 64, 27, 32, 64);
  cvtw(rw1, rw1_16, 768, 576, 576, 768);
  cvtw(rw2, rw2_16, 768, 576, 576, 768);
  cvtw(r_w1, r1_16, 128, 576, 576, 128);
  cvtw(r_w2, r2_16, 64, 288, 288, 64);
  cvtw(r_w3, r3_16, 3, 144, 160, 16);
  cvtw(k_w1, k1_16, 128, 576, 576, 128);
  cvtw(k_w2, k2_16, 400, 1152, 1152, 448);

  // conv_in: x -> F (fp32)
  conv3x3_wmma<float, float, 64, 64, 4, 2, MODE_STORE, false>
      <<<dim3(1024, 1), blk, 0, stream>>>(x, w_in16, 32, b_in, F, 4, 3, 128, 128, 64);

  // 12 residual blocks: T0 = relu(conv(relu(F))) (f16); F += conv(T0) (fp32)
  for (int i = 0; i < 12; ++i) {
    conv3x3_wmma<float, _Float16, 64, 64, 4, 2, MODE_RELU_STORE, true>
        <<<dim3(1024, 1), blk, 0, stream>>>(F, rw1_16 + (size_t)i * 36864, 576,
                                            rb1 + i * 64, T0, 4, 64, 128, 128, 64);
    conv3x3_wmma<_Float16, float, 64, 64, 4, 2, MODE_ADD, false>
        <<<dim3(1024, 1), blk, 0, stream>>>(T0, rw2_16 + (size_t)i * 36864, 576,
                                            rb2 + i * 64, F, 4, 64, 128, 128, 64);
  }

  // kernel branch: KACT = relu(conv(relu(F))); UPK = conv(KACT) [400 ch]
  conv3x3_wmma<float, _Float16, 64, 64, 4, 2, MODE_RELU_STORE, true>
      <<<dim3(1024, 2), blk, 0, stream>>>(F, k1_16, 576, k_b1, KACT, 4, 64, 128, 128, 128);
  conv3x3_wmma<_Float16, _Float16, 64, 64, 4, 2, MODE_STORE, false>
      <<<dim3(1024, 7), blk, 0, stream>>>(KACT, k2_16, 1152, k_b2, UPK, 4, 128, 128, 128, 400);

  // dynamic local filtering: out = upsampled(x, UPK)
  local_conv_us_kernel<<<dim3(65536 / 256), blk, 0, stream>>>(x, UPK, out);

  // residual branch (reuses KACT / UPK):
  conv3x3_wmma<float, _Float16, 64, 64, 4, 2, MODE_PS2_RELU, true>
      <<<dim3(1024, 2), blk, 0, stream>>>(F, r1_16, 576, r_b1, KACT, 4, 64, 128, 128, 128);
  conv3x3_wmma<_Float16, _Float16, 64, 64, 4, 2, MODE_PS2_RELU, false>
      <<<dim3(4096, 1), blk, 0, stream>>>(KACT, r2_16, 288, r_b2, UPK, 4, 32, 256, 256, 64);
  conv3x3_wmma<_Float16, float, 128, 16, 8, 1, MODE_ADD, false>
      <<<dim3(8192, 1), blk, 0, stream>>>(UPK, r3_16, 160, r_b3, out, 4, 16, 512, 512, 3);
}
